// SelfAttention_146028888406
// MI455X (gfx1250) — matrix operand order repaired and checked by reading, not compiled, not run on hardware
//
#include <hip/hip_runtime.h>

// Problem constants (from reference): B=8, C=256, OC=32, H=W=64 -> N=4096
#define B_    8
#define C_    256
#define OC_   32
#define N_    4096
#define NT_   (N_/16)    // 256 j/i tiles of 16
#define NC32_ (N_/32)    // 128 i-chunks of 32 (one WMMA K step)
#define CT_   (C_/16)    // 16 c-tiles of the 256-wide value dim

typedef __attribute__((ext_vector_type(16))) _Float16 v16h;
typedef __attribute__((ext_vector_type(8)))  float    v8f;

union F16x16 { unsigned u[8]; uint4 q[2]; v16h v; };

// ---------------------------------------------------------------------------
// Kernel 1a: Q,K projections -> f16, written directly into WMMA layouts.
//   Qlay: B-operand layout per 16-i tile  (lane = col i%16 + ((c>>3)&1 ? 16 : 0),
//          VGPR v packs K=c pairs)
//   Klay: A-operand layout per 16-j tile  (lane = row j%16 (+16 for K-half),
//          VGPR v packs K=c pairs per ISA 16-bit A 16x32 table)
// ---------------------------------------------------------------------------
__global__ void __launch_bounds__(256)
proj_qk(const float* __restrict__ x, const float* __restrict__ Wf,
        const float* __restrict__ Wg,
        _Float16* __restrict__ Qlay, _Float16* __restrict__ Klay) {
  unsigned tid = blockIdx.x * blockDim.x + threadIdx.x;
  unsigned i  = tid & (N_ - 1);
  unsigned oc = (tid >> 12) & (OC_ - 1);
  unsigned b  = tid >> 17;              // 4096*32 = 2^17
  if (b >= B_) return;
  const float* xp = x + (size_t)b * C_ * N_ + i;
  const float* wf = Wf + (size_t)oc * C_;
  const float* wg = Wg + (size_t)oc * C_;
  float q = 0.f, k = 0.f;
  for (int c = 0; c < C_; ++c) {
    float xv = xp[(size_t)c * N_];
    q += wf[c] * xv;
    k += wg[c] * xv;
  }
  { // Q -> B layout: per ISA table (lanes 0-15 carry K=0..7,16..23; lanes 16-31 carry K=8..15,24..31)
    unsigned it = i >> 4, n = i & 15;
    unsigned lane = n + (((oc >> 3) & 1) ? 16 : 0);
    unsigned v = ((oc >> 4) << 2) + ((oc & 7) >> 1);
    Qlay[((((size_t)b * NT_ + it) * 32 + lane) * 8 + v) * 2 + (oc & 1)] = (_Float16)q;
  }
  { // K -> A layout (16x32, 16-bit): per ISA table
    unsigned jt = i >> 4, m = i & 15;
    unsigned lane, v, cc = oc;
    if      (cc <  8) { lane = m;      v = cc >> 1; }
    else if (cc < 16) { lane = m + 16; v = (cc - 8) >> 1; }
    else if (cc < 24) { lane = m;      v = 4 + ((cc - 16) >> 1); }
    else              { lane = m + 16; v = 4 + ((cc - 24) >> 1); }
    Klay[((((size_t)b * NT_ + jt) * 32 + lane) * 8 + v) * 2 + (cc & 1)] = (_Float16)k;
  }
}

// ---------------------------------------------------------------------------
// Kernel 1b: V projection -> f16 in B-operand layout per (i-chunk32, c-tile16)
//   B[K=i%32][n=d%16] = v[b][d][i].  One chunk (16 KB) is contiguous so it can
//   be staged to LDS with async b128 copies in the attention kernel.
// ---------------------------------------------------------------------------
__global__ void __launch_bounds__(256)
proj_v(const float* __restrict__ x, const float* __restrict__ Wh,
       _Float16* __restrict__ Vlay) {
  unsigned tid = blockIdx.x * blockDim.x + threadIdx.x;
  unsigned i = tid & (N_ - 1);
  unsigned d = (tid >> 12) & (C_ - 1);
  unsigned b = tid >> 20;               // 4096*256 = 2^20
  if (b >= B_) return;
  const float* xp = x + (size_t)b * C_ * N_ + i;
  const float* wh = Wh + (size_t)d * C_;
  float acc = 0.f;
  for (int c = 0; c < C_; ++c) acc += wh[c] * xp[(size_t)c * N_];
  unsigned ic = i >> 5, kk = i & 31, ct = d >> 4, n = d & 15;
  unsigned lane = n + ((kk < 16) ? 0 : 16);
  unsigned vv = (kk & 15) >> 1;
  Vlay[(((((size_t)b * NC32_ + ic) * CT_ + ct) * 32 + lane) * 8 + vv) * 2 + (kk & 1)]
      = (_Float16)acc;
}

// ---------------------------------------------------------------------------
// Kernel 2a: softmax statistics, two-phase (max pass, then exp-sum pass).
// One wave per 16-row j-tile; S tiles via v_wmma_f32_16x16x32_f16.
// Phase 1 is pure v_max (dual-issue); phase 2 is 1 exp per element.
// ---------------------------------------------------------------------------
__global__ void __launch_bounds__(256)
attn_stats(const unsigned* __restrict__ Qlay, const unsigned* __restrict__ Klay,
           float* __restrict__ Mrow, float* __restrict__ Lrow) {
  unsigned lane = threadIdx.x & 31, w = threadIdx.x >> 5;
  unsigned jt = blockIdx.x * 8 + w;
  unsigned b  = blockIdx.y;

  F16x16 A;
  const uint4* kp = (const uint4*)(Klay + (((size_t)b * NT_ + jt) * 32 + lane) * 8);
  A.q[0] = kp[0]; A.q[1] = kp[1];

  float mx[8];
#pragma unroll
  for (int r = 0; r < 8; ++r) mx[r] = -1e30f;

  const unsigned* qbase = Qlay + (size_t)b * NT_ * 32 * 8 + lane * 8;
  // --- phase 1: row max ---
  for (unsigned it = 0; it < NT_; ++it) {
    F16x16 Bq;
    const uint4* qp = (const uint4*)(qbase + (size_t)it * 32 * 8);
    Bq.q[0] = qp[0]; Bq.q[1] = qp[1];
    v8f s = {0.f, 0.f, 0.f, 0.f, 0.f, 0.f, 0.f, 0.f};
    s = __builtin_amdgcn_wmma_f32_16x16x32_f16(false, A.v, false, Bq.v,
                                               (short)0, s, false, false);
#pragma unroll
    for (int r = 0; r < 8; ++r) mx[r] = fmaxf(mx[r], s[r]);
  }
#pragma unroll
  for (int mask = 1; mask < 16; mask <<= 1) {
#pragma unroll
    for (int r = 0; r < 8; ++r) mx[r] = fmaxf(mx[r], __shfl_xor(mx[r], mask, 32));
  }

  // --- phase 2: sum of exp(s - m) with fixed m ---
  float l[8];
#pragma unroll
  for (int r = 0; r < 8; ++r) l[r] = 0.f;
  for (unsigned it = 0; it < NT_; ++it) {
    F16x16 Bq;
    const uint4* qp = (const uint4*)(qbase + (size_t)it * 32 * 8);
    Bq.q[0] = qp[0]; Bq.q[1] = qp[1];
    v8f s = {0.f, 0.f, 0.f, 0.f, 0.f, 0.f, 0.f, 0.f};
    s = __builtin_amdgcn_wmma_f32_16x16x32_f16(false, A.v, false, Bq.v,
                                               (short)0, s, false, false);
#pragma unroll
    for (int r = 0; r < 8; ++r) l[r] += __expf(s[r] - mx[r]);
  }
#pragma unroll
  for (int mask = 1; mask < 16; mask <<= 1) {
#pragma unroll
    for (int r = 0; r < 8; ++r) l[r] += __shfl_xor(l[r], mask, 32);
  }

  if ((lane & 15) == 0) {
    unsigned jbase = jt * 16 + ((lane >= 16) ? 8 : 0);
#pragma unroll
    for (int r = 0; r < 8; ++r) {
      Mrow[(size_t)b * N_ + jbase + r] = mx[r];
      Lrow[(size_t)b * N_ + jbase + r] = l[r];
    }
  }
}

// ---------------------------------------------------------------------------
// Kernel 2b: recompute S, P=exp(S-m) (f16), O += P * V^T (WMMA), epilogue
// out = gamma * O / l + x.
//   * V chunks (16 KB, shared by all 8 waves) are double-buffered in LDS via
//     global_load_async_to_lds_b128 (ASYNCcnt) -> 8x fewer VMEM requests.
//   * P transposed D-layout -> A-layout through wave-private LDS (DS in-order
//     within a wave, no barrier needed for that part).
// ---------------------------------------------------------------------------
__global__ void __launch_bounds__(256)
attn_out(const float* __restrict__ x, const float* __restrict__ gamma_p,
         const unsigned* __restrict__ Qlay, const unsigned* __restrict__ Klay,
         const _Float16* __restrict__ Vlay,
         const float* __restrict__ Mrow, const float* __restrict__ Lrow,
         float* __restrict__ out) {
  __shared__ _Float16 Pl[8][16][32];            // 8 KB: per-wave P spill
  __shared__ uint4    Vbuf[2][CT_ * 32 * 2];    // 2 x 16 KB: V chunk staging

  unsigned tid  = threadIdx.x;
  unsigned lane = tid & 31, w = tid >> 5;
  unsigned jt = blockIdx.x * 8 + w;
  unsigned b  = blockIdx.y;
  float gamma = gamma_p[0];

  F16x16 A;
  const uint4* kp = (const uint4*)(Klay + (((size_t)b * NT_ + jt) * 32 + lane) * 8);
  A.q[0] = kp[0]; A.q[1] = kp[1];

  unsigned half = (lane >= 16) ? 8 : 0;
  unsigned n16  = lane & 15;
  unsigned jbase = jt * 16 + half;
  float mrow[8], linv[8];
#pragma unroll
  for (int r = 0; r < 8; ++r) {
    mrow[r] = Mrow[(size_t)b * N_ + jbase + r];
    linv[r] = 1.0f / Lrow[(size_t)b * N_ + jbase + r];
  }

  v8f acc[CT_];
#pragma unroll
  for (int ct = 0; ct < CT_; ++ct) acc[ct] = (v8f){0.f,0.f,0.f,0.f,0.f,0.f,0.f,0.f};

  unsigned* Pw32 = (unsigned*)&Pl[w][0][0];
  const unsigned* qbase = Qlay + (size_t)b * NT_ * 32 * 8 + lane * 8;

  // flat base address of this batch's V layout (16 KB contiguous per chunk)
  unsigned long long vflat = (unsigned long long)(uintptr_t)Vlay
                           + (unsigned long long)b * NC32_ * 16384ull;
  unsigned ldsv0 = (unsigned)(uintptr_t)&Vbuf[0][0];
  unsigned ldsv1 = (unsigned)(uintptr_t)&Vbuf[1][0];

  // prologue: stage chunk 0 into buffer 0 (64 B per thread, 4x b128 async)
#pragma unroll
  for (int kq = 0; kq < 4; ++kq) {
    unsigned off = tid * 64u + (unsigned)kq * 16u;
    unsigned long long ga = vflat + off;
    unsigned ld = ldsv0 + off;
    asm volatile("global_load_async_to_lds_b128 %0, %1, off"
                 :: "v"(ld), "v"(ga) : "memory");
  }
  asm volatile("s_wait_asynccnt 0x0" ::: "memory");
  __syncthreads();

  for (unsigned ic = 0; ic < NC32_; ++ic) {
    unsigned cur = ic & 1u;
    // prefetch next chunk into the other buffer
    if (ic + 1 < NC32_) {
      unsigned long long gbase = vflat + (unsigned long long)(ic + 1) * 16384ull;
      unsigned ldst = (cur ? ldsv0 : ldsv1);
#pragma unroll
      for (int kq = 0; kq < 4; ++kq) {
        unsigned off = tid * 64u + (unsigned)kq * 16u;
        unsigned long long ga = gbase + off;
        unsigned ld = ldst + off;
        asm volatile("global_load_async_to_lds_b128 %0, %1, off"
                     :: "v"(ld), "v"(ga) : "memory");
      }
    }

    // --- scores for two 16-wide i-subtiles, exp, spill f16 P to LDS ---
#pragma unroll
    for (int sub = 0; sub < 2; ++sub) {
      unsigned it = ic * 2 + sub;
      F16x16 Bq;
      const uint4* qp = (const uint4*)(qbase + (size_t)it * 32 * 8);
      Bq.q[0] = qp[0]; Bq.q[1] = qp[1];
      v8f s = {0.f, 0.f, 0.f, 0.f, 0.f, 0.f, 0.f, 0.f};
      s = __builtin_amdgcn_wmma_f32_16x16x32_f16(false, A.v, false, Bq.v,
                                                 (short)0, s, false, false);
#pragma unroll
      for (int r = 0; r < 8; ++r) {
        float p = __expf(s[r] - mrow[r]);        // 1/l applied in epilogue
        Pl[w][half + r][sub * 16 + n16] = (_Float16)p;
      }
    }
    // --- re-read P in A-operand layout (16 rows x 32 K) ---
    F16x16 Ap;
#pragma unroll
    for (int vv = 0; vv < 8; ++vv) {
      unsigned pidx = (lane < 16) ? (unsigned)vv
                                  : 8u + vv;
      Ap.u[vv] = Pw32[n16 * 16 + pidx];
    }
    // --- O += P * V^T over all 16 c-tiles (V tiles from LDS) ---
#pragma unroll
    for (int ct = 0; ct < CT_; ++ct) {
      F16x16 Bv;
      Bv.q[0] = Vbuf[cur][(ct * 32 + lane) * 2 + 0];
      Bv.q[1] = Vbuf[cur][(ct * 32 + lane) * 2 + 1];
      acc[ct] = __builtin_amdgcn_wmma_f32_16x16x32_f16(false, Ap.v, false, Bv.v,
                                                       (short)0, acc[ct], false, false);
    }

    // next buffer must be fully landed (own asyncs done + all waves arrived)
    asm volatile("s_wait_asynccnt 0x0" ::: "memory");
    __syncthreads();
  }

  // --- epilogue: out = gamma * O/l + x ---
#pragma unroll
  for (int ct = 0; ct < CT_; ++ct) {
    unsigned c = ct * 16 + n16;
#pragma unroll
    for (int r = 0; r < 8; ++r) {
      unsigned j = jbase + r;
      size_t idx = ((size_t)b * C_ + c) * N_ + j;
      out[idx] = gamma * (acc[ct][r] * linv[r]) + x[idx];
    }
  }
}

// ---------------------------------------------------------------------------
extern "C" void kernel_launch(void* const* d_in, const int* in_sizes, int n_in,
                              void* d_out, int out_size, void* d_ws, size_t ws_size,
                              hipStream_t stream) {
  (void)in_sizes; (void)n_in; (void)out_size; (void)ws_size;
  const float* x     = (const float*)d_in[0];
  const float* Wf    = (const float*)d_in[1];
  const float* Wg    = (const float*)d_in[2];
  const float* Wh    = (const float*)d_in[3];
  const float* gamma = (const float*)d_in[4];
  float* out = (float*)d_out;

  // workspace layout (~20.3 MB total)
  char* ws = (char*)d_ws;
  _Float16* Qlay = (_Float16*)(ws);                       // 2 MB
  _Float16* Klay = (_Float16*)(ws + (size_t)(2u << 20));  // 2 MB
  _Float16* Vlay = (_Float16*)(ws + (size_t)(4u << 20));  // 16 MB
  float* Mrow = (float*)(ws + (size_t)(20u << 20));       // 128 KB
  float* Lrow = (float*)(ws + (size_t)(20u << 20) + (size_t)B_ * N_ * sizeof(float));

  proj_qk<<<dim3((B_ * OC_ * N_) / 256), 256, 0, stream>>>(x, Wf, Wg, Qlay, Klay);
  proj_v <<<dim3((B_ * C_  * N_) / 256), 256, 0, stream>>>(x, Wh, Vlay);
  attn_stats<<<dim3(NT_ / 8, B_), 256, 0, stream>>>(
      (const unsigned*)Qlay, (const unsigned*)Klay, Mrow, Lrow);
  attn_out<<<dim3(NT_ / 8, B_), 256, 0, stream>>>(
      x, gamma, (const unsigned*)Qlay, (const unsigned*)Klay,
      Vlay, Mrow, Lrow, out);
}
